// Omics1_65627100283412
// MI455X (gfx1250) — compile-verified, matrix operand order checked by reading, not run on hardware
//
#include <hip/hip_runtime.h>
#include <hip/hip_bf16.h>

typedef __attribute__((ext_vector_type(16))) __bf16 v16bf;
typedef __attribute__((ext_vector_type(8)))  float  v8f;

#define N_NODES 50000
#define IN_FEAT 2000
#define OUT_FEAT 64
#define N_EDGES 800000
#define ROW_TILES 3125          // 50000 / 16
#define KT_ENC 63               // ceil(2000/32)
#define NT_ENC 4                // 64 / 16
#define KT_DEC 2                // 64 / 32
#define NT_DEC 125              // 2000 / 16
#define WAVES_PER_BLOCK 8

// ---------- helpers ----------

// Packed fp32x2 -> bf16x2 (round-to-nearest-even) in a single VALU op.
// The clang builtin __builtin_amdgcn_cvt_pk_bf16_f32 is not declared on this
// toolchain (probed via __has_builtin), but the instruction exists in the BF16
// datapath; emit it directly.
__device__ __forceinline__ unsigned bf16pack(float lo, float hi) {
    unsigned r;
    asm("v_cvt_pk_bf16_f32 %0, %1, %2" : "=v"(r) : "v"(lo), "v"(hi));
    return r;
}

// k-offset of packed dword d (0..7) for the 16-bit A/B fragment layout
__device__ __forceinline__ int frag_kbase(int d, int h) {
    return (d < 4) ? (h * 8 + 2 * d) : (16 + h * 8 + 2 * (d - 4));
}

// Load one 16x32 bf16 A fragment for row `row` starting at column kk (bound K).
__device__ __forceinline__ v16bf load_a_frag(const float* __restrict__ row,
                                             int kk, int K, int h) {
    union { unsigned u[8]; v16bf v; } fr;
    if (kk + 32 <= K) {
        const float4* p0 = reinterpret_cast<const float4*>(row + kk + h * 8);
        const float4* p1 = reinterpret_cast<const float4*>(row + kk + 16 + h * 8);
        float4 x0 = p0[0], x1 = p0[1];
        float4 y0 = p1[0], y1 = p1[1];
        fr.u[0] = bf16pack(x0.x, x0.y); fr.u[1] = bf16pack(x0.z, x0.w);
        fr.u[2] = bf16pack(x1.x, x1.y); fr.u[3] = bf16pack(x1.z, x1.w);
        fr.u[4] = bf16pack(y0.x, y0.y); fr.u[5] = bf16pack(y0.z, y0.w);
        fr.u[6] = bf16pack(y1.x, y1.y); fr.u[7] = bf16pack(y1.z, y1.w);
    } else {
#pragma unroll
        for (int d = 0; d < 8; ++d) {
            int k0 = kk + frag_kbase(d, h);
            float lo = (k0     < K) ? row[k0]     : 0.0f;
            float hi = (k0 + 1 < K) ? row[k0 + 1] : 0.0f;
            fr.u[d] = bf16pack(lo, hi);
        }
    }
    return fr.v;
}

__device__ __forceinline__ v16bf load_b_frag(const unsigned* p) {
    union { uint4 q[2]; v16bf v; } fr;
    fr.q[0] = reinterpret_cast<const uint4*>(p)[0];   // ds_load_b128
    fr.q[1] = reinterpret_cast<const uint4*>(p)[1];
    return fr.v;
}

__device__ __forceinline__ v8f wmma_bf16(v16bf a, v16bf b, v8f c) {
    return __builtin_amdgcn_wmma_f32_16x16x32_bf16(
        false, a, false, b, (short)0, c, false, false);
}

__device__ __forceinline__ void gatomic_add_f32(float* p, float v) {
    asm volatile("global_atomic_add_f32 %0, %1, off"
                 :: "v"(p), "v"(v) : "memory");
}

// ---------- encoder GEMM: xenc[50000x64] = feat[50000x2000] @ w_enc[2000x64] ----------

__global__ void gemm_enc(const float* __restrict__ feat,
                         const float* __restrict__ w,
                         float* __restrict__ xenc) {
    extern __shared__ unsigned smem[];   // [KT_ENC][NT_ENC][32][8] packed bf16 pairs
    const int tid = threadIdx.x;

    // Cooperative pack of w_enc into B-fragment layout (252 KB LDS).
    const int TOT = KT_ENC * NT_ENC * 32 * 8;   // 64512 dwords
    for (int i = tid; i < TOT; i += blockDim.x) {
        int d    = i & 7;
        int lane = (i >> 3) & 31;
        int nt   = (i >> 8) & 3;
        int kt   = i >> 10;
        int n = nt * 16 + (lane & 15);
        int h = lane >> 4;
        int k0 = kt * 32 + frag_kbase(d, h);
        float lo = (k0     < IN_FEAT) ? w[k0 * OUT_FEAT + n]       : 0.0f;
        float hi = (k0 + 1 < IN_FEAT) ? w[(k0 + 1) * OUT_FEAT + n] : 0.0f;
        smem[i] = bf16pack(lo, hi);
    }
    __syncthreads();

    const int wave = tid >> 5;
    const int lane = tid & 31;
    const int rt = blockIdx.x * WAVES_PER_BLOCK + wave;
    if (rt >= ROW_TILES) return;

    const int row0 = rt * 16;
    const int h = lane >> 4;
    const float* arow = feat + (long)(row0 + (lane & 15)) * IN_FEAT;

    v8f acc0 = {}, acc1 = {}, acc2 = {}, acc3 = {};
    for (int kt = 0; kt < KT_ENC; ++kt) {
        // Issue all 8 ds_load_b128 as one clause so the DS pipe runs ahead of
        // the WMMA pipe (only 2 waves/SIMD at this LDS footprint -> need ILP).
        const unsigned* bb = smem + ((kt * NT_ENC) * 32 + lane) * 8;
        v16bf b0 = load_b_frag(bb + 0 * 256);
        v16bf b1 = load_b_frag(bb + 1 * 256);
        v16bf b2 = load_b_frag(bb + 2 * 256);
        v16bf b3 = load_b_frag(bb + 3 * 256);

        // keep the 400MB feat stream ahead of the WMMA pipe
        __builtin_prefetch(arow + kt * 32 + 64, 0, 0);   // global_prefetch_b8
        v16bf a = load_a_frag(arow, kt * 32, IN_FEAT, h);

        acc0 = wmma_bf16(a, b0, acc0);
        acc1 = wmma_bf16(a, b1, acc1);
        acc2 = wmma_bf16(a, b2, acc2);
        acc3 = wmma_bf16(a, b3, acc3);
    }

#pragma unroll
    for (int r = 0; r < 8; ++r) {
        float* dst = xenc + (long)(row0 + h * 8 + r) * OUT_FEAT + (lane & 15);
        dst[0]  = acc0[r];
        dst[16] = acc1[r];
        dst[32] = acc2[r];
        dst[48] = acc3[r];
    }
}

// ---------- spmm: out[rows[e]][j] += vals[e] * x[cols[e]][j], j in 0..63 ----------

__global__ void spmm64(const int* __restrict__ rows,
                       const int* __restrict__ cols,
                       const float* __restrict__ vals,
                       const float* __restrict__ x,
                       float* outbuf) {
    long t = (long)blockIdx.x * blockDim.x + threadIdx.x;
    if (t >= (long)N_EDGES * OUT_FEAT) return;
    int e = (int)(t >> 6);
    int j = (int)(t & 63);
    float v = vals[e] * x[(long)cols[e] * OUT_FEAT + j];
    gatomic_add_f32(outbuf + (long)rows[e] * OUT_FEAT + j, v);
}

// ---------- decoder GEMM: recon[50000x2000] = y[50000x64] @ w_dec[64x2000] ----------

__global__ void gemm_dec(const float* __restrict__ y,
                         const float* __restrict__ w,
                         float* __restrict__ out) {
    extern __shared__ unsigned smem[];   // [KT_DEC][NT_DEC][32][8] packed bf16 pairs
    const int tid = threadIdx.x;

    const int TOT = KT_DEC * NT_DEC * 32 * 8;   // 64000 dwords (250 KB)
    for (int i = tid; i < TOT; i += blockDim.x) {
        int d    = i & 7;
        int lane = (i >> 3) & 31;
        int g    = i >> 8;          // kt*125 + nt
        int nt   = g % NT_DEC;
        int kt   = g / NT_DEC;
        int n = nt * 16 + (lane & 15);
        int h = lane >> 4;
        int k0 = kt * 32 + frag_kbase(d, h);   // always < 64
        smem[i] = bf16pack(w[k0 * IN_FEAT + n], w[(k0 + 1) * IN_FEAT + n]);
    }
    __syncthreads();

    const int wave = tid >> 5;
    const int lane = tid & 31;
    const int rt = blockIdx.x * WAVES_PER_BLOCK + wave;
    if (rt >= ROW_TILES) return;

    const int row0 = rt * 16;
    const int h = lane >> 4;
    const float* arow = y + (long)(row0 + (lane & 15)) * OUT_FEAT;

    // K = 64: both A fragments stay resident in VGPRs across all n-tiles.
    v16bf a0 = load_a_frag(arow, 0,  OUT_FEAT, h);
    v16bf a1 = load_a_frag(arow, 32, OUT_FEAT, h);

    for (int nt = 0; nt < NT_DEC; ++nt) {
        const unsigned* b0p = smem + ((0 * NT_DEC + nt) * 32 + lane) * 8;
        const unsigned* b1p = smem + ((1 * NT_DEC + nt) * 32 + lane) * 8;
        v16bf b0 = load_b_frag(b0p);
        v16bf b1 = load_b_frag(b1p);
        v8f acc = {};
        acc = wmma_bf16(a0, b0, acc);
        acc = wmma_bf16(a1, b1, acc);
        const int n = nt * 16 + (lane & 15);
#pragma unroll
        for (int r = 0; r < 8; ++r) {
            out[(long)(row0 + h * 8 + r) * IN_FEAT + n] = acc[r];
        }
    }
}

// ---------- host ----------

extern "C" void kernel_launch(void* const* d_in, const int* in_sizes, int n_in,
                              void* d_out, int out_size, void* d_ws, size_t ws_size,
                              hipStream_t stream) {
    const float* feat  = (const float*)d_in[0];
    const int*   rows  = (const int*)  d_in[1];
    const int*   cols  = (const int*)  d_in[2];
    const float* vals  = (const float*)d_in[3];
    const float* w_enc = (const float*)d_in[4];
    const float* w_dec = (const float*)d_in[5];

    float* x_latent = (float*)d_out;                              // [N, 64]
    float* x_recon  = x_latent + (size_t)N_NODES * OUT_FEAT;      // [N, 2000]

    float* xenc = (float*)d_ws;                                   // [N, 64]
    float* yacc = xenc + (size_t)N_NODES * OUT_FEAT;              // [N, 64]

    const size_t latent_bytes = (size_t)N_NODES * OUT_FEAT * sizeof(float);
    hipMemsetAsync(x_latent, 0, latent_bytes, stream);
    hipMemsetAsync(yacc,     0, latent_bytes, stream);

    const int gemm_blocks = (ROW_TILES + WAVES_PER_BLOCK - 1) / WAVES_PER_BLOCK; // 391
    const size_t lds_enc = (size_t)KT_ENC * NT_ENC * 32 * 8 * 4;  // 258048 B
    const size_t lds_dec = (size_t)KT_DEC * NT_DEC * 32 * 8 * 4;  // 256000 B

    // 1) xenc = feat @ w_enc
    gemm_enc<<<gemm_blocks, 256, lds_enc, stream>>>(feat, w_enc, xenc);

    // 2) x_latent = adj @ xenc
    const int spmm_blocks = (int)(((long)N_EDGES * OUT_FEAT + 255) / 256);  // 200000
    spmm64<<<spmm_blocks, 256, 0, stream>>>(rows, cols, vals, xenc, x_latent);

    // 3) yacc = adj @ x_latent   (associativity: adj @ (X W) = (adj X) W)
    spmm64<<<spmm_blocks, 256, 0, stream>>>(rows, cols, vals, x_latent, yacc);

    // 4) x_recon = yacc @ w_dec
    gemm_dec<<<gemm_blocks, 256, lds_dec, stream>>>(yacc, w_dec, x_recon);
}
